// SchNet_51608327029068
// MI455X (gfx1250) — matrix operand order repaired and checked by reading, not compiled, not run on hardware
//
#include <hip/hip_runtime.h>

#define NNODES  100000
#define NEDGES  1600000
#define HDIM    128
#define FWDIM   128
#define NGAUSS  50
#define KG      64
#define NGRAPH  64
#define NINTER  6
#define CUTOFF_F 10.0f
#define LOG2_F   0.69314718055994531f
#define PI_F     3.14159265358979323f

typedef _Float16 h8   __attribute__((ext_vector_type(8)));
typedef _Float16 v16h __attribute__((ext_vector_type(16)));
typedef float    v8f  __attribute__((ext_vector_type(8)));
typedef int      i4v  __attribute__((ext_vector_type(4)));

#define WMMA_F16(a,b,c) __builtin_amdgcn_wmma_f32_16x16x32_f16(false,(a),false,(b),(short)0,(c),false,false)

#if defined(__has_builtin)
#if __has_builtin(__builtin_amdgcn_global_load_async_to_lds_b128)
#define HAVE_ASYNC_LDS 1
#endif
#endif

typedef __attribute__((address_space(1))) i4v* g_i4p;   // global int4*
typedef __attribute__((address_space(3))) i4v* l_i4p;   // LDS int4*

// Stage 16-byte chunks global->LDS. Uses CDNA5 async-to-LDS (ASYNCcnt) when the
// builtin exists; otherwise plain register staging.
__device__ __forceinline__ void stage_to_lds_b128(const _Float16* g, _Float16* l,
                                                  int nchunks, int tid, int nthreads) {
#ifdef HAVE_ASYNC_LDS
  for (int i = tid; i < nchunks; i += nthreads)
    __builtin_amdgcn_global_load_async_to_lds_b128(
        (g_i4p)(g + (size_t)i * 8), (l_i4p)(l + (size_t)i * 8), 0, 0);
#else
  for (int i = tid; i < nchunks; i += nthreads)
    ((h8*)l)[i] = ((const h8*)g)[i];
#endif
}

__device__ __forceinline__ float sspf(float x) {
  // softplus(x) - log(2), numerically stable
  float ax = __builtin_fabsf(x);
  return fmaxf(x, 0.f) + log1pf(__expf(-ax)) - LOG2_F;
}

// ---- WMMA fragment builders -------------------------------------------------
// A 16x32 f16: lanes 0-15 row=lane K={k0+0..7, k0+16..23}; lanes 16-31 row=lane-16
// K={k0+8..15, k0+24..31}. Element 2v,2v+1 of v16h live in VGPR v.
__device__ __forceinline__ v16h frag_a_f16(const _Float16* base, int row0, int ld,
                                           int k0, int lane) {
  int r  = row0 + (lane & 15);
  int kb = k0 + ((lane & 16) ? 8 : 0);
  const _Float16* p = base + (size_t)r * ld + kb;
  h8 lo = *(const h8*)(p);
  h8 hi = *(const h8*)(p + 16);
  v16h o;
#pragma unroll
  for (int i = 0; i < 8; i++) { o[i] = lo[i]; o[i + 8] = hi[i]; }
  return o;
}

// Same A pattern but sourced from row-major f32 with on-the-fly f16 convert.
__device__ __forceinline__ v16h frag_a_f32(const float* base, int row0, int ld,
                                           int k0, int lane) {
  int r  = row0 + (lane & 15);
  int kb = k0 + ((lane & 16) ? 8 : 0);
  const float* p = base + (size_t)r * ld + kb;
  v16h o;
#pragma unroll
  for (int i = 0; i < 8; i++) {
    o[i]     = (_Float16)p[i];
    o[i + 8] = (_Float16)p[16 + i];
  }
  return o;
}

// B 32x16 f16 from weights stored transposed [Nout, K]:
// lanes 0-15 col=lane K={k0+0..7, k0+16..23}; lanes 16-31 col=lane-16 K={k0+8..15, k0+24..31}.
__device__ __forceinline__ v16h frag_b(const _Float16* wt, int n0, int ldk,
                                       int k0, int lane) {
  int c  = n0 + (lane & 15);
  int kb = k0 + ((lane & 16) ? 8 : 0);
  const _Float16* p = wt + (size_t)c * ldk + kb;
  h8 lo = *(const h8*)(p);
  h8 hi = *(const h8*)(p + 16);
  v16h o;
#pragma unroll
  for (int i = 0; i < 8; i++) { o[i] = lo[i]; o[i + 8] = hi[i]; }
  return o;
}

// ---- prep kernels -----------------------------------------------------------
__global__ void edge_geom_kernel(const float* __restrict__ pos,
                                 const int* __restrict__ ei,
                                 float* __restrict__ ew, float* __restrict__ cw) {
  int e = blockIdx.x * blockDim.x + threadIdx.x;
  if (e >= NEDGES) return;
  int r = ei[e], c = ei[NEDGES + e];
  float dx = pos[3 * r + 0] - pos[3 * c + 0];
  float dy = pos[3 * r + 1] - pos[3 * c + 1];
  float dz = pos[3 * r + 2] - pos[3 * c + 2];
  float d = sqrtf(dx * dx + dy * dy + dz * dz);
  ew[e] = d;
  cw[e] = 0.5f * (__cosf(d * (PI_F / CUTOFF_F)) + 1.0f);
}

__global__ void node_init_kernel(const int* __restrict__ z,
                                 const float* __restrict__ emb,
                                 float* __restrict__ h) {
  long i = (long)blockIdx.x * blockDim.x + threadIdx.x;
  if (i >= (long)NNODES * HDIM) return;
  int n = (int)(i >> 7), c = (int)(i & 127);
  h[i] = emb[(size_t)z[n] * HDIM + c];
}

__global__ void zero_f32(float* __restrict__ p, long n) {
  long i = (long)blockIdx.x * blockDim.x + threadIdx.x;
  if (i < n) p[i] = 0.f;
}

// src is [K, Nout] row-major f32; dst is [Nout, Kpad] f16 (transposed, K padded w/ 0)
__global__ void transpose_w_kernel(const float* __restrict__ src,
                                   _Float16* __restrict__ dst,
                                   int K, int Nout, int Kpad) {
  int i = blockIdx.x * blockDim.x + threadIdx.x;
  if (i >= Nout * Kpad) return;
  int n = i / Kpad, k = i % Kpad;
  dst[i] = (k < K) ? (_Float16)src[(size_t)k * Nout + n] : (_Float16)0.f;
}

// ---- xl = h @ conv_w1 (WMMA, f32 out) --------------------------------------
__global__ void __launch_bounds__(256)
lin1_kernel(const float* __restrict__ hbuf, const _Float16* __restrict__ wt,
            float* __restrict__ xl) {
  int lane = threadIdx.x & 31, wv = threadIdx.x >> 5;
  int mt = blockIdx.x * 8 + wv;                    // wave-uniform guard (EXEC all-1)
  if (mt >= NNODES / 16) return;
  int r0 = mt * 16;
  v16h a[4];
#pragma unroll
  for (int ks = 0; ks < 4; ks++) a[ks] = frag_a_f32(hbuf, r0, HDIM, ks * 32, lane);
  int mb = (lane & 16) ? 8 : 0;
  int col_lo = lane & 15;
#pragma unroll
  for (int nt = 0; nt < 8; nt++) {
    v8f acc = {};
#pragma unroll
    for (int ks = 0; ks < 4; ks++)
      acc = WMMA_F16(a[ks], frag_b(wt, nt * 16, HDIM, ks * 32, lane), acc);
    int col = nt * 16 + col_lo;
#pragma unroll
    for (int v = 0; v < 8; v++)
      xl[(size_t)(r0 + mb + v) * FWDIM + col] = acc[v];
  }
}

// ---- fused edge pipeline: gaussians -> MLP(WMMA x2) -> *C -> gather -> scatter
// Weights staged once per block into LDS (async-to-LDS when available), so each
// WMMA pulls B operands via ds_load instead of re-reading global per tile.
__global__ void __launch_bounds__(256)
edge_filter_kernel(const float* __restrict__ ew, const float* __restrict__ cw,
                   const int* __restrict__ ei,
                   const _Float16* __restrict__ w1t, const float* __restrict__ b1,
                   const _Float16* __restrict__ w2t, const float* __restrict__ b2,
                   const float* __restrict__ xl, float* __restrict__ agg) {
  __shared__ __align__(16) _Float16 s_w1[FWDIM * KG];        // 16 KB [128][64]
  __shared__ __align__(16) _Float16 s_w2[FWDIM * FWDIM];     // 32 KB [128][128]
  __shared__ __align__(16) _Float16 lds_hid[8 * 16 * 128];   // 32 KB per-wave tiles
  int tid = threadIdx.x;

  stage_to_lds_b128(w1t, s_w1, FWDIM * KG / 8, tid, 256);
  stage_to_lds_b128(w2t, s_w2, FWDIM * FWDIM / 8, tid, 256);
#ifdef HAVE_ASYNC_LDS
  asm volatile("s_wait_asynccnt 0" ::: "memory");
#endif
  __syncthreads();

  int lane = tid & 31, wv = tid >> 5;
  int mt = blockIdx.x * 8 + wv;                    // wave-uniform (EXEC all-1)
  if (mt >= NEDGES / 16) return;
  int e0 = mt * 16;

  // On-the-fly gaussian smearing for this lane's A-row (no [E,G] array in HBM)
  float ew_a = ew[e0 + (lane & 15)];
  const float gstep = CUTOFF_F / (NGAUSS - 1);
  const float gco = -0.5f / (gstep * gstep);
  v16h a1[2];
#pragma unroll
  for (int ks = 0; ks < 2; ks++) {
    int kb = ks * 32 + ((lane & 16) ? 8 : 0);
    v16h o;
#pragma unroll
    for (int i = 0; i < 8; i++) {
      int g0 = kb + i, g1 = kb + 16 + i;
      float d0 = ew_a - (float)g0 * gstep;
      float d1 = ew_a - (float)g1 * gstep;
      o[i]     = (g0 < NGAUSS) ? (_Float16)__expf(gco * d0 * d0) : (_Float16)0.f;
      o[i + 8] = (g1 < NGAUSS) ? (_Float16)__expf(gco * d1 * d1) : (_Float16)0.f;
    }
    a1[ks] = o;
  }

  _Float16* myl = &lds_hid[wv * 16 * 128];
  int col_lo = lane & 15;
  int mb = (lane & 16) ? 8 : 0;

  // GEMM1: [16,64]x[64,128] -> ssp -> LDS (D-layout to A-layout relayout)
#pragma unroll
  for (int nt = 0; nt < 8; nt++) {
    v8f acc = {};
    acc = WMMA_F16(a1[0], frag_b(s_w1, nt * 16, KG, 0, lane), acc);
    acc = WMMA_F16(a1[1], frag_b(s_w1, nt * 16, KG, 32, lane), acc);
    int col = nt * 16 + col_lo;
    float bb = b1[col];
#pragma unroll
    for (int v = 0; v < 8; v++)
      myl[(mb + v) * 128 + col] = (_Float16)sspf(acc[v] + bb);
  }
  asm volatile("s_wait_dscnt 0" ::: "memory");   // same-wave LDS RAW across lanes

  v16h a2[4];
#pragma unroll
  for (int ks = 0; ks < 4; ks++) a2[ks] = frag_a_f16(myl, 0, 128, ks * 32, lane);

  // Per-lane edge metadata for its D-layout rows
  int rn[8], cn[8]; float cwv[8];
#pragma unroll
  for (int v = 0; v < 8; v++) {
    int e = e0 + mb + v;
    rn[v] = ei[e];
    cn[v] = ei[NEDGES + e];
    cwv[v] = cw[e];
  }

  // GEMM2: [16,128]x[128,128]; then W*=C, msg = xl[row]*W, scatter-add to agg[col]
#pragma unroll
  for (int nt = 0; nt < 8; nt++) {
    v8f acc = {};
#pragma unroll
    for (int ks = 0; ks < 4; ks++)
      acc = WMMA_F16(a2[ks], frag_b(s_w2, nt * 16, FWDIM, ks * 32, lane), acc);
    int col = nt * 16 + col_lo;
    float bb = b2[col];
#pragma unroll
    for (int v = 0; v < 8; v++) {
      float W = (acc[v] + bb) * cwv[v];
      float msg = xl[(size_t)rn[v] * FWDIM + col] * W;
      atomicAdd(&agg[(size_t)cn[v] * FWDIM + col], msg);
    }
  }
}

// ---- node update: x = ssp(agg@W2+b)@Wint+b ; h += x ------------------------
__global__ void __launch_bounds__(256)
node_update_kernel(const float* __restrict__ agg,
                   const _Float16* __restrict__ cw2t, const float* __restrict__ cb2,
                   const _Float16* __restrict__ intwt, const float* __restrict__ intb,
                   float* __restrict__ hbuf) {
  __shared__ __align__(16) _Float16 lds_x[8 * 16 * 128];
  int lane = threadIdx.x & 31, wv = threadIdx.x >> 5;
  int mt = blockIdx.x * 8 + wv;
  if (mt >= NNODES / 16) return;
  int r0 = mt * 16;
  v16h a[4];
#pragma unroll
  for (int ks = 0; ks < 4; ks++) a[ks] = frag_a_f32(agg, r0, FWDIM, ks * 32, lane);
  _Float16* myl = &lds_x[wv * 16 * 128];
  int col_lo = lane & 15;
  int mb = (lane & 16) ? 8 : 0;
#pragma unroll
  for (int nt = 0; nt < 8; nt++) {
    v8f acc = {};
#pragma unroll
    for (int ks = 0; ks < 4; ks++)
      acc = WMMA_F16(a[ks], frag_b(cw2t, nt * 16, FWDIM, ks * 32, lane), acc);
    int col = nt * 16 + col_lo;
    float bb = cb2[col];
#pragma unroll
    for (int v = 0; v < 8; v++)
      myl[(mb + v) * 128 + col] = (_Float16)sspf(acc[v] + bb);
  }
  asm volatile("s_wait_dscnt 0" ::: "memory");
  v16h a2[4];
#pragma unroll
  for (int ks = 0; ks < 4; ks++) a2[ks] = frag_a_f16(myl, 0, 128, ks * 32, lane);
#pragma unroll
  for (int nt = 0; nt < 8; nt++) {
    v8f acc = {};
#pragma unroll
    for (int ks = 0; ks < 4; ks++)
      acc = WMMA_F16(a2[ks], frag_b(intwt, nt * 16, HDIM, ks * 32, lane), acc);
    int col = nt * 16 + col_lo;
    float bb = intb[col];
#pragma unroll
    for (int v = 0; v < 8; v++) {
      size_t idx = (size_t)(r0 + mb + v) * HDIM + col;
      hbuf[idx] += acc[v] + bb;                 // residual, tile owns rows
    }
  }
}

// ---- readout head: ssp(h@W1+b1)@W2+b2, per-graph atomic sum ----------------
__global__ void __launch_bounds__(256)
readout_kernel(const float* __restrict__ hbuf, const int* __restrict__ batch,
               const float* __restrict__ ow1, const float* __restrict__ ob1,
               const float* __restrict__ ow2, const float* __restrict__ ob2,
               float* __restrict__ out) {
  __shared__ float lw1[128 * 64];
  __shared__ float lw2[64];
  __shared__ float lb1[64];
  for (int t = threadIdx.x; t < 128 * 64; t += 256) lw1[t] = ow1[t];
  for (int t = threadIdx.x; t < 64; t += 256) { lw2[t] = ow2[t]; lb1[t] = ob1[t]; }
  __syncthreads();
  int n = blockIdx.x * 256 + threadIdx.x;
  if (n >= NNODES) return;
  float acc[64];
#pragma unroll
  for (int j = 0; j < 64; j++) acc[j] = 0.f;
  for (int k = 0; k < 128; k++) {
    float hk = hbuf[(size_t)n * 128 + k];
#pragma unroll
    for (int j = 0; j < 64; j++) acc[j] = fmaf(hk, lw1[k * 64 + j], acc[j]);
  }
  float val = ob2[0];
#pragma unroll
  for (int j = 0; j < 64; j++) val += sspf(acc[j] + lb1[j]) * lw2[j];
  atomicAdd(&out[batch[n]], val);
}

// ---- host -------------------------------------------------------------------
extern "C" void kernel_launch(void* const* d_in, const int* in_sizes, int n_in,
                              void* d_out, int out_size, void* d_ws, size_t ws_size,
                              hipStream_t stream) {
  (void)in_sizes; (void)n_in; (void)out_size; (void)ws_size;
  const int*   z        = (const int*)d_in[0];
  const float* pos      = (const float*)d_in[1];
  const int*   ei       = (const int*)d_in[2];
  const int*   batch    = (const int*)d_in[3];
  const float* emb      = (const float*)d_in[4];
  const float* mlp_w1   = (const float*)d_in[5];
  const float* mlp_b1   = (const float*)d_in[6];
  const float* mlp_w2   = (const float*)d_in[7];
  const float* mlp_b2   = (const float*)d_in[8];
  const float* conv_w1  = (const float*)d_in[9];
  const float* conv_w2  = (const float*)d_in[10];
  const float* conv_b2  = (const float*)d_in[11];
  const float* int_w    = (const float*)d_in[12];
  const float* int_b    = (const float*)d_in[13];
  const float* out_w1   = (const float*)d_in[14];
  const float* out_b1   = (const float*)d_in[15];
  const float* out_w2   = (const float*)d_in[16];
  const float* out_b2   = (const float*)d_in[17];
  float* out = (float*)d_out;

  char* ws = (char*)d_ws;
  size_t off = 0;
  auto take = [&](size_t bytes) -> char* {
    char* p = ws + off;
    off += (bytes + 255) & ~(size_t)255;
    return p;
  };
  float*    ew    = (float*)take((size_t)NEDGES * 4);
  float*    cwb   = (float*)take((size_t)NEDGES * 4);
  float*    hbuf  = (float*)take((size_t)NNODES * HDIM * 4);
  float*    xl    = (float*)take((size_t)NNODES * FWDIM * 4);
  float*    agg   = (float*)take((size_t)NNODES * FWDIM * 4);
  _Float16* w1t   = (_Float16*)take((size_t)FWDIM * KG * 2);
  _Float16* w2t   = (_Float16*)take((size_t)FWDIM * FWDIM * 2);
  _Float16* cw1t  = (_Float16*)take((size_t)FWDIM * HDIM * 2);
  _Float16* cw2t  = (_Float16*)take((size_t)HDIM * FWDIM * 2);
  _Float16* intwt = (_Float16*)take((size_t)HDIM * HDIM * 2);

  edge_geom_kernel<<<(NEDGES + 255) / 256, 256, 0, stream>>>(pos, ei, ew, cwb);
  node_init_kernel<<<((long)NNODES * HDIM + 255) / 256, 256, 0, stream>>>(z, emb, hbuf);

  const int node_tiles = NNODES / 16;               // 6250
  const int node_blocks = (node_tiles + 7) / 8;     // 782
  const int edge_blocks = (NEDGES / 16) / 8;        // 12500

  for (int i = 0; i < NINTER; i++) {
    transpose_w_kernel<<<(128 * 64 + 255) / 256, 256, 0, stream>>>(
        mlp_w1 + (size_t)i * NGAUSS * FWDIM, w1t, NGAUSS, FWDIM, KG);
    transpose_w_kernel<<<(128 * 128 + 255) / 256, 256, 0, stream>>>(
        mlp_w2 + (size_t)i * FWDIM * FWDIM, w2t, FWDIM, FWDIM, FWDIM);
    transpose_w_kernel<<<(128 * 128 + 255) / 256, 256, 0, stream>>>(
        conv_w1 + (size_t)i * HDIM * FWDIM, cw1t, HDIM, FWDIM, HDIM);
    transpose_w_kernel<<<(128 * 128 + 255) / 256, 256, 0, stream>>>(
        conv_w2 + (size_t)i * FWDIM * HDIM, cw2t, FWDIM, HDIM, FWDIM);
    transpose_w_kernel<<<(128 * 128 + 255) / 256, 256, 0, stream>>>(
        int_w + (size_t)i * HDIM * HDIM, intwt, HDIM, HDIM, HDIM);

    zero_f32<<<((long)NNODES * FWDIM + 255) / 256, 256, 0, stream>>>(
        agg, (long)NNODES * FWDIM);
    lin1_kernel<<<node_blocks, 256, 0, stream>>>(hbuf, cw1t, xl);
    edge_filter_kernel<<<edge_blocks, 256, 0, stream>>>(
        ew, cwb, ei, w1t, mlp_b1 + (size_t)i * FWDIM, w2t,
        mlp_b2 + (size_t)i * FWDIM, xl, agg);
    node_update_kernel<<<node_blocks, 256, 0, stream>>>(
        agg, cw2t, conv_b2 + (size_t)i * HDIM, intwt, int_b + (size_t)i * HDIM, hbuf);
  }

  zero_f32<<<1, 64, 0, stream>>>(out, NGRAPH);
  readout_kernel<<<(NNODES + 255) / 256, 256, 0, stream>>>(
      hbuf, batch, out_w1, out_b1, out_w2, out_b2, out);
}
